// SparseTransformer_14628658610294
// MI455X (gfx1250) — matrix operand order repaired and checked by reading, not compiled, not run on hardware
//
#include <hip/hip_runtime.h>
#include <hip/hip_bf16.h>

#define B_DIM 16384
#define N_DIM 1024
#define M_DIM 4096
#define A_DIM 1024

typedef __attribute__((ext_vector_type(16))) _Float16 v16h;
typedef __attribute__((ext_vector_type(8)))  _Float16 v8h;
typedef __attribute__((ext_vector_type(8)))  float    v8f;
typedef __attribute__((ext_vector_type(4)))  float    v4f;

#define WMMA_F16(a, b, c) \
  __builtin_amdgcn_wmma_f32_16x16x32_f16(false, (a), false, (b), (short)0, (c), false, false)

// ---------------- fragment loaders ----------------
// A-operand (16x32 f16, M x K): lane (l&15)=row; K chunks {kb..kb+7, kb+16..kb+23}, kb = k0 + 8*(l>=16)
__device__ __forceinline__ v16h loadA_f16(const _Float16* __restrict__ p, size_t ldk,
                                          int m, int k0, int lane) {
  const _Float16* b = p + (size_t)m * ldk + k0 + ((lane & 16) ? 8 : 0);
  v8h c0 = *(const v8h*)(b);
  v8h c1 = *(const v8h*)(b + 16);
  v16h r;
#pragma unroll
  for (int i = 0; i < 8; ++i) { r[i] = c0[i]; r[i + 8] = c1[i]; }
  return r;
}

__device__ __forceinline__ v16h loadA_f32(const float* __restrict__ p, size_t ldk,
                                          int m, int k0, int lane) {
  const float* b = p + (size_t)m * ldk + k0 + ((lane & 16) ? 8 : 0);
  v4f c0 = *(const v4f*)(b);
  v4f c1 = *(const v4f*)(b + 4);
  v4f c2 = *(const v4f*)(b + 16);
  v4f c3 = *(const v4f*)(b + 20);
  v16h r;
#pragma unroll
  for (int i = 0; i < 4; ++i) {
    r[i]      = (_Float16)c0[i];
    r[i + 4]  = (_Float16)c1[i];
    r[i + 8]  = (_Float16)c2[i];
    r[i + 12] = (_Float16)c3[i];
  }
  return r;
}

// B-operand (32x16 f16, K x N), stored as [N, K] row-major: lane (l&15)=col;
// K chunks {kb..kb+7, kb+16..kb+23}, kb = k0 + 8*(l>=16)
__device__ __forceinline__ v16h loadB_f16(const _Float16* __restrict__ p, size_t ldk,
                                          int n, int k0, int lane) {
  const _Float16* b = p + (size_t)n * ldk + k0 + ((lane & 16) ? 8 : 0);
  v8h c0 = *(const v8h*)(b);
  v8h c1 = *(const v8h*)(b + 16);
  v16h r;
#pragma unroll
  for (int i = 0; i < 8; ++i) { r[i] = c0[i]; r[i + 8] = c1[i]; }
  return r;
}

// ---------------- input convert: x -> out_x (copy) + x16; wq -> wq16 ----------------
__global__ __launch_bounds__(256) void cvt_inputs(
    const float* __restrict__ x, const float* __restrict__ wq,
    float* __restrict__ out_x, _Float16* __restrict__ x16,
    _Float16* __restrict__ wq16) {
  const size_t nx = (size_t)B_DIM * N_DIM;
  size_t i = ((size_t)blockIdx.x * 256 + threadIdx.x) * 8;
  const float* src;
  _Float16* dst;
  float* cpy = nullptr;
  size_t off;
  if (i < nx) { src = x; dst = x16; cpy = out_x; off = i; }
  else        { src = wq; dst = wq16; off = i - nx; }
  v4f a = *(const v4f*)(src + off);
  v4f b = *(const v4f*)(src + off + 4);
  if (cpy) {
    *(v4f*)(cpy + off)     = a;
    *(v4f*)(cpy + off + 4) = b;
  }
  v8h h;
#pragma unroll
  for (int j = 0; j < 4; ++j) { h[j] = (_Float16)a[j]; h[j + 4] = (_Float16)b[j]; }
  *(v8h*)(dst + off) = h;
}

// ---------------- k / v prep: relu + layernorm ----------------
__global__ __launch_bounds__(256) void prep_kv(
    const float* __restrict__ wk, const float* __restrict__ wv,
    const float* __restrict__ gk, const float* __restrict__ betak,
    const float* __restrict__ gv, const float* __restrict__ betav,
    _Float16* __restrict__ k16, float* __restrict__ out_v,
    _Float16* __restrict__ vT16) {
  __shared__ float red0[9], red1[9];
  const int tid  = threadIdx.x;
  const int lane = tid & 31, wave = tid >> 5;
  const bool isV = blockIdx.x >= M_DIM;
  const int r    = isV ? (blockIdx.x - M_DIM) : blockIdx.x;
  const float* src = (isV ? wv : wk) + (size_t)r * 1024;

  float vals[4], s = 0.f, s2 = 0.f;
#pragma unroll
  for (int i = 0; i < 4; ++i) {
    float v = fmaxf(src[tid + i * 256], 0.f);
    vals[i] = v; s += v; s2 += v * v;
  }
#pragma unroll
  for (int off = 16; off > 0; off >>= 1) {
    s  += __shfl_xor(s, off, 32);
    s2 += __shfl_xor(s2, off, 32);
  }
  if (lane == 0) { red0[wave] = s; red1[wave] = s2; }
  __syncthreads();
  if (tid == 0) {
    float t0 = 0.f, t1 = 0.f;
    for (int w = 0; w < 8; ++w) { t0 += red0[w]; t1 += red1[w]; }
    red0[8] = t0; red1[8] = t1;
  }
  __syncthreads();
  float mu  = red0[8] * (1.f / 1024.f);
  float var = red1[8] * (1.f / 1024.f) - mu * mu;
  float rs  = rsqrtf(var + 1e-5f);
#pragma unroll
  for (int i = 0; i < 4; ++i) {
    int c = tid + i * 256;
    if (!isV) {
      float o = (vals[i] - mu) * rs * gk[c] + betak[c];
      k16[(size_t)r * 1024 + c] = (_Float16)o;
    } else {
      float o = (vals[i] - mu) * rs * gv[c] + betav[c];
      out_v[(size_t)r * 1024 + c] = o;
      vT16[(size_t)c * M_DIM + r] = (_Float16)o;
    }
  }
}

// ---------------- GEMM1: q = LN(relu(x @ wq^T + bq)), fused epilogue ----------------
// block = 16 rows x 1024 cols (full row -> fused layernorm). 8 waves, each 16x128.
// two-stage software pipeline on K
__global__ __launch_bounds__(256) void gemm1_q(
    const _Float16* __restrict__ x16, const _Float16* __restrict__ wq16,
    const float* __restrict__ bq, const float* __restrict__ gq,
    const float* __restrict__ betaq, _Float16* __restrict__ q16) {
  __shared__ float tile[16 * 1024];
  const int tid  = threadIdx.x;
  const int lane = tid & 31, wave = tid >> 5;
  const int m0   = blockIdx.x * 16;
  const int n0   = wave * 128;

  v8f acc[8];
#pragma unroll
  for (int t = 0; t < 8; ++t) acc[t] = v8f{};

  const int mrow = m0 + (lane & 15);
  v16h a0, a1, b0[8], b1[8];
  a0 = loadA_f16(x16, N_DIM, mrow, 0, lane);
#pragma unroll
  for (int t = 0; t < 8; ++t)
    b0[t] = loadB_f16(wq16, N_DIM, n0 + t * 16 + (lane & 15), 0, lane);

  for (int k0 = 0; k0 < N_DIM; k0 += 64) {
    const int k1 = k0 + 32;
    const int k2 = (k0 + 64) & (N_DIM - 1);  // wrapped prefetch (last iter discarded)
    // prefetch stage 1
    a1 = loadA_f16(x16, N_DIM, mrow, k1, lane);
#pragma unroll
    for (int t = 0; t < 8; ++t)
      b1[t] = loadB_f16(wq16, N_DIM, n0 + t * 16 + (lane & 15), k1, lane);
    // compute stage 0
#pragma unroll
    for (int t = 0; t < 8; ++t) acc[t] = WMMA_F16(a0, b0[t], acc[t]);
    // prefetch stage 0 for next iteration
    a0 = loadA_f16(x16, N_DIM, mrow, k2, lane);
#pragma unroll
    for (int t = 0; t < 8; ++t)
      b0[t] = loadB_f16(wq16, N_DIM, n0 + t * 16 + (lane & 15), k2, lane);
    // compute stage 1
#pragma unroll
    for (int t = 0; t < 8; ++t) acc[t] = WMMA_F16(a1, b1[t], acc[t]);
  }
  // bias + relu into LDS tile
  const int mhalf = (lane >> 4) << 3;
#pragma unroll
  for (int t = 0; t < 8; ++t) {
    int n = n0 + t * 16 + (lane & 15);
    float bias = bq[n];
#pragma unroll
    for (int e = 0; e < 8; ++e) {
      float v = acc[t][e] + bias;
      tile[(e + mhalf) * 1024 + n] = fmaxf(v, 0.f);
    }
  }
  __syncthreads();
  // layernorm: each 16-thread group owns one of the 16 rows
  const int row = tid >> 4, j = tid & 15;
  float s = 0.f, s2 = 0.f;
  for (int c = j; c < 1024; c += 16) {
    float v = tile[row * 1024 + c];
    s += v; s2 += v * v;
  }
#pragma unroll
  for (int off = 8; off > 0; off >>= 1) {
    s  += __shfl_xor(s, off, 16);
    s2 += __shfl_xor(s2, off, 16);
  }
  float mu  = s * (1.f / 1024.f);
  float var = s2 * (1.f / 1024.f) - mu * mu;
  float rs  = rsqrtf(var + 1e-5f);
  for (int c = j; c < 1024; c += 16) {
    float v = tile[row * 1024 + c];
    float o = (v - mu) * rs * gq[c] + betaq[c];
    q16[(size_t)(m0 + row) * 1024 + c] = (_Float16)o;
  }
}

// ---------------- GEMM2: scores = (q @ k^T) * 1/sqrt(A) ----------------
// block tile 128(M) x 128(N); waves 4(M) x 2(N); wave tile 32 x 64
// two-stage software pipeline on K
__global__ __launch_bounds__(256) void gemm2_scores(
    const _Float16* __restrict__ q16, const _Float16* __restrict__ k16,
    float* __restrict__ scores) {
  const int tid  = threadIdx.x;
  const int lane = tid & 31, wave = tid >> 5;
  const int wm = wave & 3, wn = wave >> 2;
  const int m0 = blockIdx.y * 128 + wm * 32;
  const int n0 = blockIdx.x * 128 + wn * 64;

  v8f acc[2][4];
#pragma unroll
  for (int mt = 0; mt < 2; ++mt)
#pragma unroll
    for (int nt = 0; nt < 4; ++nt) acc[mt][nt] = v8f{};

  v16h a0[2], b0[4], a1[2], b1[4];
#pragma unroll
  for (int mt = 0; mt < 2; ++mt)
    a0[mt] = loadA_f16(q16, A_DIM, m0 + mt * 16 + (lane & 15), 0, lane);
#pragma unroll
  for (int nt = 0; nt < 4; ++nt)
    b0[nt] = loadB_f16(k16, A_DIM, n0 + nt * 16 + (lane & 15), 0, lane);

  for (int k0 = 0; k0 < A_DIM; k0 += 64) {
    const int k1 = k0 + 32;
    const int k2 = (k0 + 64) & (A_DIM - 1);
#pragma unroll
    for (int mt = 0; mt < 2; ++mt)
      a1[mt] = loadA_f16(q16, A_DIM, m0 + mt * 16 + (lane & 15), k1, lane);
#pragma unroll
    for (int nt = 0; nt < 4; ++nt)
      b1[nt] = loadB_f16(k16, A_DIM, n0 + nt * 16 + (lane & 15), k1, lane);
#pragma unroll
    for (int mt = 0; mt < 2; ++mt)
#pragma unroll
      for (int nt = 0; nt < 4; ++nt)
        acc[mt][nt] = WMMA_F16(a0[mt], b0[nt], acc[mt][nt]);
#pragma unroll
    for (int mt = 0; mt < 2; ++mt)
      a0[mt] = loadA_f16(q16, A_DIM, m0 + mt * 16 + (lane & 15), k2, lane);
#pragma unroll
    for (int nt = 0; nt < 4; ++nt)
      b0[nt] = loadB_f16(k16, A_DIM, n0 + nt * 16 + (lane & 15), k2, lane);
#pragma unroll
    for (int mt = 0; mt < 2; ++mt)
#pragma unroll
      for (int nt = 0; nt < 4; ++nt)
        acc[mt][nt] = WMMA_F16(a1[mt], b1[nt], acc[mt][nt]);
  }
  const float scale = 0.03125f;  // 1/sqrt(1024)
  const int mhalf = (lane >> 4) << 3;
#pragma unroll
  for (int mt = 0; mt < 2; ++mt)
#pragma unroll
    for (int nt = 0; nt < 4; ++nt) {
      int n = n0 + nt * 16 + (lane & 15);
#pragma unroll
      for (int e = 0; e < 8; ++e) {
        int m = m0 + mt * 16 + mhalf + e;
        scores[(size_t)m * M_DIM + n] = acc[mt][nt][e] * scale;
      }
    }
}

// ---------------- sparsemax, in place, one block per row of 4096 ----------------
// optionally emits an f16 copy of f for GEMM3's A operand
__global__ __launch_bounds__(256) void sparsemax_inplace(
    float* __restrict__ f, _Float16* __restrict__ f16out) {
  __shared__ float srow[M_DIM];
  __shared__ float red0[9], red1[9];
  const int tid  = threadIdx.x;
  const int lane = tid & 31, wave = tid >> 5;
  float* row = f + (size_t)blockIdx.x * M_DIM;

  float lmax = -3.4e38f;
#pragma unroll
  for (int i = 0; i < 16; ++i) {
    int c = tid + i * 256;
    float s = row[c];
    s = (s < -1e10f) ? 0.f : s;
    srow[c] = s;
    lmax = fmaxf(lmax, s);
  }
#pragma unroll
  for (int off = 16; off > 0; off >>= 1) lmax = fmaxf(lmax, __shfl_xor(lmax, off, 32));
  if (lane == 0) red0[wave] = lmax;
  __syncthreads();
  if (tid == 0) {
    float m = red0[0];
    for (int w = 1; w < 8; ++w) m = fmaxf(m, red0[w]);
    red0[8] = m;
  }
  __syncthreads();
  const float smax = red0[8];

  // bisection on g(tau) = sum(max(s - tau, 0)); g(smax-1) >= 1 >= g(smax)
  float lo = smax - 1.f, hi = smax;
  for (int it = 0; it < 30; ++it) {
    float tau = 0.5f * (lo + hi);
    float g = 0.f;
#pragma unroll
    for (int i = 0; i < 16; ++i) g += fmaxf(srow[tid + i * 256] - tau, 0.f);
#pragma unroll
    for (int off = 16; off > 0; off >>= 1) g += __shfl_xor(g, off, 32);
    if (lane == 0) red0[wave] = g;
    __syncthreads();
    if (tid == 0) {
      float t = 0.f;
      for (int w = 0; w < 8; ++w) t += red0[w];
      red0[8] = t;
    }
    __syncthreads();
    if (red0[8] > 1.f) lo = tau; else hi = tau;
    __syncthreads();
  }
  // exact refinement: tau = (sum_{s>tau0} s - 1) / |{s>tau0}|
  float tau0 = 0.5f * (lo + hi);
  float cnt = 0.f, ssum = 0.f;
#pragma unroll
  for (int i = 0; i < 16; ++i) {
    float s = srow[tid + i * 256];
    if (s > tau0) { cnt += 1.f; ssum += s; }
  }
#pragma unroll
  for (int off = 16; off > 0; off >>= 1) {
    cnt  += __shfl_xor(cnt, off, 32);
    ssum += __shfl_xor(ssum, off, 32);
  }
  if (lane == 0) { red0[wave] = ssum; red1[wave] = cnt; }
  __syncthreads();
  if (tid == 0) {
    float t0 = 0.f, t1 = 0.f;
    for (int w = 0; w < 8; ++w) { t0 += red0[w]; t1 += red1[w]; }
    red0[8] = t0; red1[8] = t1;
  }
  __syncthreads();
  const float tau = (red0[8] - 1.f) / fmaxf(red1[8], 1.f);
#pragma unroll
  for (int i = 0; i < 16; ++i) {
    int c = tid + i * 256;
    float o = fmaxf(srow[c] - tau, 0.f);
    row[c] = o;
    if (f16out) f16out[(size_t)blockIdx.x * M_DIM + c] = (_Float16)o;
  }
}

// ---------------- GEMM3: x_hat = f @ v  (v^T pre-staged f16) ----------------
// block tile 128(M) x 128(N); waves 4(M) x 2(N); wave tile 32 x 64; pipelined
__global__ __launch_bounds__(256) void gemm3_xhat_f16(
    const _Float16* __restrict__ f16, const _Float16* __restrict__ vT16,
    float* __restrict__ xhat) {
  const int tid  = threadIdx.x;
  const int lane = tid & 31, wave = tid >> 5;
  const int wm = wave & 3, wn = wave >> 2;
  const int m0 = blockIdx.y * 128 + wm * 32;
  const int n0 = blockIdx.x * 128 + wn * 64;

  v8f acc[2][4];
#pragma unroll
  for (int mt = 0; mt < 2; ++mt)
#pragma unroll
    for (int nt = 0; nt < 4; ++nt) acc[mt][nt] = v8f{};

  v16h a0[2], b0[4], a1[2], b1[4];
#pragma unroll
  for (int mt = 0; mt < 2; ++mt)
    a0[mt] = loadA_f16(f16, M_DIM, m0 + mt * 16 + (lane & 15), 0, lane);
#pragma unroll
  for (int nt = 0; nt < 4; ++nt)
    b0[nt] = loadB_f16(vT16, M_DIM, n0 + nt * 16 + (lane & 15), 0, lane);

  for (int k0 = 0; k0 < M_DIM; k0 += 64) {
    const int k1 = k0 + 32;
    const int k2 = (k0 + 64) & (M_DIM - 1);
#pragma unroll
    for (int mt = 0; mt < 2; ++mt)
      a1[mt] = loadA_f16(f16, M_DIM, m0 + mt * 16 + (lane & 15), k1, lane);
#pragma unroll
    for (int nt = 0; nt < 4; ++nt)
      b1[nt] = loadB_f16(vT16, M_DIM, n0 + nt * 16 + (lane & 15), k1, lane);
#pragma unroll
    for (int mt = 0; mt < 2; ++mt)
#pragma unroll
      for (int nt = 0; nt < 4; ++nt)
        acc[mt][nt] = WMMA_F16(a0[mt], b0[nt], acc[mt][nt]);
#pragma unroll
    for (int mt = 0; mt < 2; ++mt)
      a0[mt] = loadA_f16(f16, M_DIM, m0 + mt * 16 + (lane & 15), k2, lane);
#pragma unroll
    for (int nt = 0; nt < 4; ++nt)
      b0[nt] = loadB_f16(vT16, M_DIM, n0 + nt * 16 + (lane & 15), k2, lane);
#pragma unroll
    for (int mt = 0; mt < 2; ++mt)
#pragma unroll
      for (int nt = 0; nt < 4; ++nt)
        acc[mt][nt] = WMMA_F16(a1[mt], b1[nt], acc[mt][nt]);
  }
  const int mhalf = (lane >> 4) << 3;
#pragma unroll
  for (int mt = 0; mt < 2; ++mt)
#pragma unroll
    for (int nt = 0; nt < 4; ++nt) {
      int n = n0 + nt * 16 + (lane & 15);
#pragma unroll
      for (int e = 0; e < 8; ++e) {
        int m = m0 + mt * 16 + mhalf + e;
        xhat[(size_t)m * N_DIM + n] = acc[mt][nt][e];
      }
    }
}

// fallback when ws is too small for an f16 copy of f: convert f32 f on load
__global__ __launch_bounds__(256) void gemm3_xhat_f32(
    const float* __restrict__ f, const _Float16* __restrict__ vT16,
    float* __restrict__ xhat) {
  const int tid  = threadIdx.x;
  const int lane = tid & 31, wave = tid >> 5;
  const int wm = wave & 3, wn = wave >> 2;
  const int m0 = blockIdx.y * 128 + wm * 32;
  const int n0 = blockIdx.x * 128 + wn * 64;

  v8f acc[2][4];
#pragma unroll
  for (int mt = 0; mt < 2; ++mt)
#pragma unroll
    for (int nt = 0; nt < 4; ++nt) acc[mt][nt] = v8f{};

  for (int k0 = 0; k0 < M_DIM; k0 += 32) {
    v16h a[2], b[4];
#pragma unroll
    for (int mt = 0; mt < 2; ++mt)
      a[mt] = loadA_f32(f, M_DIM, m0 + mt * 16 + (lane & 15), k0, lane);
#pragma unroll
    for (int nt = 0; nt < 4; ++nt)
      b[nt] = loadB_f16(vT16, M_DIM, n0 + nt * 16 + (lane & 15), k0, lane);
#pragma unroll
    for (int mt = 0; mt < 2; ++mt)
#pragma unroll
      for (int nt = 0; nt < 4; ++nt)
        acc[mt][nt] = WMMA_F16(a[mt], b[nt], acc[mt][nt]);
  }
  const int mhalf = (lane >> 4) << 3;
#pragma unroll
  for (int mt = 0; mt < 2; ++mt)
#pragma unroll
    for (int nt = 0; nt < 4; ++nt) {
      int n = n0 + nt * 16 + (lane & 15);
#pragma unroll
      for (int e = 0; e < 8; ++e) {
        int m = m0 + mt * 16 + mhalf + e;
        xhat[(size_t)m * N_DIM + n] = acc[mt][nt][e];
      }
    }
}

// ---------------- launch ----------------
extern "C" void kernel_launch(void* const* d_in, const int* in_sizes, int n_in,
                              void* d_out, int out_size, void* d_ws, size_t ws_size,
                              hipStream_t stream) {
  (void)in_sizes; (void)n_in; (void)out_size;
  const float* x     = (const float*)d_in[0];
  const float* wq    = (const float*)d_in[1];
  const float* bq    = (const float*)d_in[2];
  const float* wk    = (const float*)d_in[3];
  const float* wv    = (const float*)d_in[4];
  const float* gq    = (const float*)d_in[5];
  const float* betaq = (const float*)d_in[6];
  const float* gk    = (const float*)d_in[7];
  const float* betak = (const float*)d_in[8];
  const float* gv    = (const float*)d_in[9];
  const float* betav = (const float*)d_in[10];

  float* out      = (float*)d_out;
  float* out_x    = out;                                                      // [B, N]
  float* out_xhat = out + (size_t)B_DIM * N_DIM;                              // [B, N]
  float* out_f    = out + 2 * (size_t)B_DIM * N_DIM;                          // [B, M]
  float* out_v    = out + 2 * (size_t)B_DIM * N_DIM + (size_t)B_DIM * M_DIM;  // [M, N]

  char* ws = (char*)d_ws;
  size_t o = 0;
  _Float16* x16  = (_Float16*)(ws + o); o += (size_t)B_DIM * N_DIM * 2;  // 33.5 MB
  _Float16* q16  = (_Float16*)(ws + o); o += (size_t)B_DIM * A_DIM * 2;  // 33.5 MB
  _Float16* k16  = (_Float16*)(ws + o); o += (size_t)M_DIM * A_DIM * 2;  //  8.4 MB
  _Float16* vT16 = (_Float16*)(ws + o); o += (size_t)N_DIM * M_DIM * 2;  //  8.4 MB
  _Float16* wq16 = (_Float16*)(ws + o); o += (size_t)A_DIM * N_DIM * 2;  //  2.1 MB
  const size_t f16_bytes = (size_t)B_DIM * M_DIM * 2;                    // 134 MB
  const bool use_f16f = (ws_size >= o + f16_bytes);
  _Float16* f16 = use_f16f ? (_Float16*)(ws + o) : nullptr;

  const size_t cvt_elems = (size_t)B_DIM * N_DIM + (size_t)A_DIM * N_DIM;
  cvt_inputs<<<(unsigned)(cvt_elems / (256 * 8)), 256, 0, stream>>>(
      x, wq, out_x, x16, wq16);
  prep_kv<<<2 * M_DIM, 256, 0, stream>>>(wk, wv, gk, betak, gv, betav,
                                         k16, out_v, vT16);
  gemm1_q<<<B_DIM / 16, 256, 0, stream>>>(x16, wq16, bq, gq, betaq, q16);
  gemm2_scores<<<dim3(M_DIM / 128, B_DIM / 128), 256, 0, stream>>>(q16, k16, out_f);
  sparsemax_inplace<<<B_DIM, 256, 0, stream>>>(out_f, f16);
  if (use_f16f)
    gemm3_xhat_f16<<<dim3(N_DIM / 128, B_DIM / 128), 256, 0, stream>>>(f16, vT16, out_xhat);
  else
    gemm3_xhat_f32<<<dim3(N_DIM / 128, B_DIM / 128), 256, 0, stream>>>(out_f, vT16, out_xhat);
}
